// MoE_3925600108741
// MI455X (gfx1250) — matrix operand order repaired and checked by reading, not compiled, not run on hardware
//
#include <hip/hip_runtime.h>
#include <hip/hip_bf16.h>
#include <math.h>

// ---------------------------------------------------------------------------
// MoE (GShard top-1) for MI455X / gfx1250:
//   bf16 WMMA grouped GEMMs, double-buffered async global->LDS pipeline,
//   ordered ballot-scan routing, fused combine epilogue.
// ---------------------------------------------------------------------------

typedef __bf16 bf16;
typedef __attribute__((ext_vector_type(16))) __bf16 v16bf;
typedef __attribute__((ext_vector_type(8)))  float  v8f;

#define S_TOK 16384   // B*S tokens
#define MDIM  1024
#define NEXP  8
#define FDIM  4096
#define CAP   2048    // S_TOK / NEXP

#define BM 128
#define BN 128
#define BK 64
#define LDSTRIDE (BK + 8)   // 72 bf16 per LDS row (144 B, 16B-aligned)

// ---- CDNA5 async global->LDS copy (ASYNCcnt-tracked, no VGPR round trip) ----
static __device__ __forceinline__ unsigned lds_off(const void* p) {
  // generic pointer to a __shared__ object: low 32 bits are the LDS offset
  return (unsigned)(uintptr_t)p;
}
static __device__ __forceinline__ void async_copy_b128(unsigned dst_lds,
                                                       const void* src) {
  asm volatile("global_load_async_to_lds_b128 %0, %1, off"
               :: "v"(dst_lds), "v"((unsigned long long)(uintptr_t)src)
               : "memory");
}
static __device__ __forceinline__ void wait_async0() {   // all async done
  asm volatile("s_wait_asynccnt 0x0" ::: "memory");
}
static __device__ __forceinline__ void wait_async8() {   // oldest stage done
  asm volatile("s_wait_asynccnt 0x8" ::: "memory");
}

static __device__ __forceinline__ unsigned ballot32(bool p) {
#if defined(__has_builtin)
#if __has_builtin(__builtin_amdgcn_ballot_w32)
  return __builtin_amdgcn_ballot_w32(p);
#else
  return (unsigned)__ballot(p);
#endif
#else
  return (unsigned)__ballot(p);
#endif
}

// ---------------------------------------------------------------------------
// fills
// ---------------------------------------------------------------------------
__global__ void moe_fill_u32(unsigned* __restrict__ p, size_t n, unsigned v) {
  size_t i  = (size_t)blockIdx.x * blockDim.x + threadIdx.x;
  size_t st = (size_t)gridDim.x * blockDim.x;
  for (; i < n; i += st) p[i] = v;
}

// ---------------------------------------------------------------------------
// weight pre-pass: fp32 [E][K][N] -> bf16 [E][N][K] (LDS-tiled transpose)
// ---------------------------------------------------------------------------
__global__ __launch_bounds__(256)
void moe_transpose_bf16(const float* __restrict__ src, bf16* __restrict__ dst,
                        int K, int N) {
  __shared__ float tile[32][33];
  const int e  = blockIdx.z;
  const int n0 = blockIdx.x * 32;
  const int k0 = blockIdx.y * 32;
  const int tx = threadIdx.x & 31;
  const int ty = threadIdx.x >> 5;          // 8 row-groups

  const float* s = src + ((size_t)e * K + k0) * N + n0;
#pragma unroll
  for (int r = ty; r < 32; r += 8) tile[r][tx] = s[(size_t)r * N + tx];
  __syncthreads();

  bf16* d = dst + ((size_t)e * N + n0) * K + k0;
#pragma unroll
  for (int r = ty; r < 32; r += 8) d[(size_t)r * K + tx] = (bf16)tile[tx][r];
}

// ---------------------------------------------------------------------------
// gating: one wave per token. logits = x[t,:] @ wg -> softmax -> argmax
// ---------------------------------------------------------------------------
__global__ __launch_bounds__(256)
void moe_gate(const float* __restrict__ x, const float* __restrict__ wg,
              int* __restrict__ idx, float* __restrict__ gmax,
              float* __restrict__ gall) {
  const int wid  = threadIdx.x >> 5;
  const int lane = threadIdx.x & 31;
  const int t = blockIdx.x * 8 + wid;
  if (t >= S_TOK) return;

  float acc[NEXP];
#pragma unroll
  for (int e = 0; e < NEXP; ++e) acc[e] = 0.f;

  const float* xr = x + (size_t)t * MDIM;
  for (int m = lane; m < MDIM; m += 32) {
    const float xv = xr[m];
    const float* wr = wg + (size_t)m * NEXP;
#pragma unroll
    for (int e = 0; e < NEXP; ++e) acc[e] += xv * wr[e];
  }
#pragma unroll
  for (int e = 0; e < NEXP; ++e) {
#pragma unroll
    for (int off = 16; off > 0; off >>= 1)
      acc[e] += __shfl_xor(acc[e], off, 32);
  }
  float mx = acc[0];
#pragma unroll
  for (int e = 1; e < NEXP; ++e) mx = fmaxf(mx, acc[e]);
  float sum = 0.f, g[NEXP];
#pragma unroll
  for (int e = 0; e < NEXP; ++e) { g[e] = __expf(acc[e] - mx); sum += g[e]; }
  const float inv = 1.f / sum;
  int best = 0; float bg = g[0] * inv;
#pragma unroll
  for (int e = 0; e < NEXP; ++e) {
    g[e] *= inv;
    if (g[e] > bg) { bg = g[e]; best = e; }   // strict '>' => first max (argmax)
  }
  if (lane == 0) {
    idx[t]  = best;
    gmax[t] = bg;
#pragma unroll
    for (int e = 0; e < NEXP; ++e) gall[(size_t)t * NEXP + e] = g[e];
  }
}

// ---------------------------------------------------------------------------
// ordered scan: single wave. Exact token-order ranks (cumsum(mask)-1),
// capacity drop, slot->token inverse map, deterministic l_aux.
// ---------------------------------------------------------------------------
__global__ void moe_scan(const int* __restrict__ idx, const float* __restrict__ gmax,
                         const float* __restrict__ gall,
                         int* __restrict__ loc, float* __restrict__ gfin,
                         int* __restrict__ slot, float* __restrict__ laux_out) {
  const int lane = threadIdx.x;           // 32 threads, one wave
  const unsigned ltmask = (lane == 0) ? 0u : ((1u << lane) - 1u);
  int   cnt[NEXP];
  float me[NEXP], ce[NEXP];
#pragma unroll
  for (int e = 0; e < NEXP; ++e) { cnt[e] = 0; me[e] = 0.f; ce[e] = 0.f; }

  for (int base = 0; base < S_TOK; base += 32) {
    const int t = base + lane;
    const int e = idx[t];
    const float gv = gmax[t];
    unsigned bal[NEXP];
#pragma unroll
    for (int ee = 0; ee < NEXP; ++ee) bal[ee] = ballot32(e == ee);
    const int rank = cnt[e] + __popc(bal[e] & ltmask);
    const bool keep = rank < CAP;
    loc[t]  = keep ? rank : 0;
    gfin[t] = keep ? gv : 0.f;
    if (keep) slot[e * CAP + rank] = t;
#pragma unroll
    for (int ee = 0; ee < NEXP; ++ee) cnt[ee] += __popc(bal[ee]);
#pragma unroll
    for (int ee = 0; ee < NEXP; ++ee) me[ee] += gall[(size_t)t * NEXP + ee];
    ce[e] += 1.f;
  }
#pragma unroll
  for (int ee = 0; ee < NEXP; ++ee) {
#pragma unroll
    for (int off = 16; off > 0; off >>= 1) {
      me[ee] += __shfl_xor(me[ee], off, 32);
      ce[ee] += __shfl_xor(ce[ee], off, 32);
    }
  }
  if (lane == 0) {
    const float invS = 1.f / (float)S_TOK;
    float s = 0.f;
#pragma unroll
    for (int ee = 0; ee < NEXP; ++ee) s += (me[ee] * invS) * (ce[ee] * invS);
    laux_out[0] = s * (float)NEXP;      // mean_e(me*ce) * E^2
  }
}

// ---------------------------------------------------------------------------
// dispatch: scatter kept tokens into disp[e][cap][M] as bf16
// ---------------------------------------------------------------------------
__global__ __launch_bounds__(256)
void moe_dispatch(const float* __restrict__ x, const int* __restrict__ idx,
                  const int* __restrict__ loc, const float* __restrict__ gfin,
                  bf16* __restrict__ disp) {
  const int t = blockIdx.x;
  if (gfin[t] <= 0.f) return;                  // dropped token
  const int e = idx[t], c = loc[t];
  bf16* drow = disp + ((size_t)e * CAP + c) * MDIM;
  const float* xr = x + (size_t)t * MDIM;
  for (int i = threadIdx.x; i < MDIM; i += blockDim.x)
    drow[i] = (bf16)xr[i];
}

// ---------------------------------------------------------------------------
// async-stage a BMxBK bf16 tile (row-major, ld = KD): 4 fixed 16B chunks per
// thread, no loop guard (tid < 256 guaranteed by launch config).
// 4 async instructions per wave per tile => 8 per (A,B) stage.
// ---------------------------------------------------------------------------
static __device__ __forceinline__ void stage_tile_async(
    bf16 (*lds)[LDSTRIDE], const bf16* __restrict__ g, size_t ldk,
    int row0, int k0, int tid) {
  const int rbase = tid >> 3;         // 0..31
  const int cu    = tid & 7;          // 0..7 (8 bf16 = 16 B each)
#pragma unroll
  for (int i = 0; i < 4; ++i) {
    const int r = rbase + i * 32;     // 0..127
    async_copy_b128(lds_off(&lds[r][cu * 8]),
                    g + (size_t)(row0 + r) * ldk + k0 + cu * 8);
  }
}

// compute one BK=64 stage: two K=32 subtiles, 8 WMMAs each
static __device__ __forceinline__ void wmma_stage(
    const bf16 (*Alds)[LDSTRIDE], const bf16 (*Blds)[LDSTRIDE],
    v8f acc[2][4], int waveM, int waveN, int hh, int l16) {
#pragma unroll
  for (int s = 0; s < 2; ++s) {
    const int ko = s * 32;
    union { uint4 u[2]; v16bf v; } fa[2], fb[4];
#pragma unroll
    for (int i = 0; i < 2; ++i) {
      const int r = waveM + i * 16 + l16;
      fa[i].u[0] = *(const uint4*)&Alds[r][ko + hh * 8];
      fa[i].u[1] = *(const uint4*)&Alds[r][ko + 16 + hh * 8];
    }
#pragma unroll
    for (int j = 0; j < 4; ++j) {
      const int n = waveN + j * 16 + l16;
      fb[j].u[0] = *(const uint4*)&Blds[n][ko + hh * 8];
      fb[j].u[1] = *(const uint4*)&Blds[n][ko + 16 + hh * 8];
    }
#pragma unroll
    for (int i = 0; i < 2; ++i)
#pragma unroll
      for (int j = 0; j < 4; ++j)
        acc[i][j] = __builtin_amdgcn_wmma_f32_16x16x32_bf16(
            false, fa[i].v, false, fb[j].v, (short)0, acc[i][j], false, false);
  }
}

// ---------------------------------------------------------------------------
// GEMM1: h[e] = gelu(disp[e] (2048x1024 bf16) @ w1T[e] ([4096][1024] bf16) + b1)
// Double-buffered async pipeline: prefetch stage k+1 while computing stage k.
// ---------------------------------------------------------------------------
__global__ __launch_bounds__(256)
void moe_gemm1(const bf16* __restrict__ disp, const bf16* __restrict__ w1T,
               const float* __restrict__ b1, bf16* __restrict__ h) {
  const int e  = blockIdx.z;
  const int m0 = blockIdx.y * BM;
  const int n0 = blockIdx.x * BN;
  const int KD = MDIM;
  const bf16* A = disp + (size_t)e * CAP * KD;
  const bf16* B = w1T + (size_t)e * FDIM * KD;     // [N][K]

  __shared__ __align__(16) bf16 Alds[2][BM][LDSTRIDE];
  __shared__ __align__(16) bf16 Blds[2][BN][LDSTRIDE];

  const int tid = threadIdx.x, lane = tid & 31, wid = tid >> 5;
  const int waveM = (wid & 3) * 32, waveN = (wid >> 2) * 64;
  const int hh = lane >> 4, l16 = lane & 15;

  v8f acc[2][4];
#pragma unroll
  for (int i = 0; i < 2; ++i)
#pragma unroll
    for (int j = 0; j < 4; ++j) acc[i][j] = (v8f){};

  stage_tile_async(Alds[0], A, KD, m0, 0, tid);
  stage_tile_async(Blds[0], B, KD, n0, 0, tid);

  const int NK = KD / BK;
  for (int ki = 0; ki < NK; ++ki) {
    const int cur = ki & 1;
    if (ki + 1 < NK) {                       // prefetch next stage
      stage_tile_async(Alds[cur ^ 1], A, KD, m0, (ki + 1) * BK, tid);
      stage_tile_async(Blds[cur ^ 1], B, KD, n0, (ki + 1) * BK, tid);
      wait_async8();                         // oldest 8 (stage ki) complete
    } else {
      wait_async0();
    }
    __syncthreads();
    wmma_stage(Alds[cur], Blds[cur], acc, waveM, waveN, hh, l16);
    __syncthreads();
  }

#pragma unroll
  for (int i = 0; i < 2; ++i) {
#pragma unroll
    for (int j = 0; j < 4; ++j) {
      const int col  = n0 + waveN + j * 16 + l16;
      const float bias = b1[(size_t)e * FDIM + col];
#pragma unroll
      for (int v = 0; v < 8; ++v) {
        const int row = m0 + waveM + i * 16 + hh * 8 + v;
        const float xv = acc[i][j][v] + bias;
        const float gl = 0.5f * xv * (1.f + erff(xv * 0.70710678118f));
        h[((size_t)e * CAP + row) * FDIM + col] = (bf16)gl;
      }
    }
  }
}

// ---------------------------------------------------------------------------
// GEMM2 + combine: out[token] = gate * (h[e] @ w2T[e] + b2) via slot map
// ---------------------------------------------------------------------------
__global__ __launch_bounds__(256)
void moe_gemm2(const bf16* __restrict__ h, const bf16* __restrict__ w2T,
               const float* __restrict__ b2, const int* __restrict__ slot,
               const float* __restrict__ gfin, float* __restrict__ out) {
  const int e  = blockIdx.z;
  const int m0 = blockIdx.y * BM;
  const int n0 = blockIdx.x * BN;
  const int KD = FDIM;
  const bf16* A = h   + (size_t)e * CAP * KD;
  const bf16* B = w2T + (size_t)e * MDIM * KD;     // [N][K]

  __shared__ __align__(16) bf16 Alds[2][BM][LDSTRIDE];
  __shared__ __align__(16) bf16 Blds[2][BN][LDSTRIDE];

  const int tid = threadIdx.x, lane = tid & 31, wid = tid >> 5;
  const int waveM = (wid & 3) * 32, waveN = (wid >> 2) * 64;
  const int hh = lane >> 4, l16 = lane & 15;

  v8f acc[2][4];
#pragma unroll
  for (int i = 0; i < 2; ++i)
#pragma unroll
    for (int j = 0; j < 4; ++j) acc[i][j] = (v8f){};

  stage_tile_async(Alds[0], A, KD, m0, 0, tid);
  stage_tile_async(Blds[0], B, KD, n0, 0, tid);

  const int NK = KD / BK;
  for (int ki = 0; ki < NK; ++ki) {
    const int cur = ki & 1;
    if (ki + 1 < NK) {
      stage_tile_async(Alds[cur ^ 1], A, KD, m0, (ki + 1) * BK, tid);
      stage_tile_async(Blds[cur ^ 1], B, KD, n0, (ki + 1) * BK, tid);
      wait_async8();
    } else {
      wait_async0();
    }
    __syncthreads();
    wmma_stage(Alds[cur], Blds[cur], acc, waveM, waveN, hh, l16);
    __syncthreads();
  }

#pragma unroll
  for (int i = 0; i < 2; ++i) {
#pragma unroll
    for (int j = 0; j < 4; ++j) {
      const int col  = n0 + waveN + j * 16 + l16;
      const float bias = b2[(size_t)e * MDIM + col];
#pragma unroll
      for (int v = 0; v < 8; ++v) {
        const int row = m0 + waveM + i * 16 + hh * 8 + v;   // capacity slot
        const int tok = slot[e * CAP + row];
        if (tok >= 0) {
          out[(size_t)tok * MDIM + col] = gfin[tok] * (acc[i][j][v] + bias);
        }
      }
    }
  }
}

// ---------------------------------------------------------------------------
// launch
// ---------------------------------------------------------------------------
extern "C" void kernel_launch(void* const* d_in, const int* in_sizes, int n_in,
                              void* d_out, int out_size, void* d_ws, size_t ws_size,
                              hipStream_t stream) {
  const float* x  = (const float*)d_in[0];   // [8,2048,1024]
  const float* wg = (const float*)d_in[1];   // [1024,8]
  const float* w1 = (const float*)d_in[2];   // [8,1024,4096]
  const float* b1 = (const float*)d_in[3];   // [8,4096]
  const float* w2 = (const float*)d_in[4];   // [8,4096,1024]
  const float* b2 = (const float*)d_in[5];   // [8,1024]
  float* out = (float*)d_out;                // [8,2048,1024] + 1 (l_aux)

  // workspace layout (~302 MB)
  char* ws = (char*)d_ws;
  size_t off = 0;
  bf16* disp = (bf16*)(ws + off);  off += (size_t)NEXP * CAP * MDIM * sizeof(bf16);
  bf16* hbuf = (bf16*)(ws + off);  off += (size_t)NEXP * CAP * FDIM * sizeof(bf16);
  bf16* w1T  = (bf16*)(ws + off);  off += (size_t)NEXP * FDIM * MDIM * sizeof(bf16);
  bf16* w2T  = (bf16*)(ws + off);  off += (size_t)NEXP * MDIM * FDIM * sizeof(bf16);
  float* gall = (float*)(ws + off); off += (size_t)S_TOK * NEXP * sizeof(float);
  int*   idx  = (int*)(ws + off);   off += (size_t)S_TOK * sizeof(int);
  int*   loc  = (int*)(ws + off);   off += (size_t)S_TOK * sizeof(int);
  float* gmax = (float*)(ws + off); off += (size_t)S_TOK * sizeof(float);
  float* gfin = (float*)(ws + off); off += (size_t)S_TOK * sizeof(float);
  int*   slot = (int*)(ws + off);   off += (size_t)NEXP * CAP * sizeof(int);

  // 1) zero out (incl. l_aux slot), zero disp, slot = -1
  moe_fill_u32<<<4096, 256, 0, stream>>>((unsigned*)out,
      (size_t)S_TOK * MDIM + 1, 0u);
  moe_fill_u32<<<4096, 256, 0, stream>>>((unsigned*)disp,
      (size_t)NEXP * CAP * MDIM / 2, 0u);
  moe_fill_u32<<<64, 256, 0, stream>>>((unsigned*)slot,
      (size_t)NEXP * CAP, 0xFFFFFFFFu);

  // 2) weight pre-pass: transpose + convert to bf16 [E][N][K]
  moe_transpose_bf16<<<dim3(FDIM / 32, MDIM / 32, NEXP), 256, 0, stream>>>(
      w1, w1T, MDIM, FDIM);
  moe_transpose_bf16<<<dim3(MDIM / 32, FDIM / 32, NEXP), 256, 0, stream>>>(
      w2, w2T, FDIM, MDIM);

  // 3) gating (one wave per token)
  moe_gate<<<S_TOK / 8, 256, 0, stream>>>(x, wg, idx, gmax, gall);

  // 4) ordered scan + l_aux (single wave, deterministic)
  moe_scan<<<1, 32, 0, stream>>>(idx, gmax, gall, loc, gfin, slot,
                                 out + (size_t)S_TOK * MDIM);

  // 5) dispatch kept tokens to bf16
  moe_dispatch<<<S_TOK, 256, 0, stream>>>(x, idx, loc, gfin, disp);

  // 6) grouped GEMM1 + GeLU   (per expert: 2048x4096x1024)
  moe_gemm1<<<dim3(FDIM / BN, CAP / BM, NEXP), 256, 0, stream>>>(disp, w1T, b1, hbuf);

  // 7) grouped GEMM2 + fused combine (per expert: 2048x1024x4096)
  moe_gemm2<<<dim3(MDIM / BN, CAP / BM, NEXP), 256, 0, stream>>>(hbuf, w2T, b2,
                                                                 slot, gfin, out);
}
